// Context2Query_52475910423101
// MI455X (gfx1250) — matrix operand order repaired and checked by reading, not compiled, not run on hardware
//
#include <hip/hip_runtime.h>

typedef __attribute__((ext_vector_type(16))) _Float16 v16h;
typedef __attribute__((ext_vector_type(8)))  _Float16 v8h;
typedef __attribute__((ext_vector_type(8)))  float    v8f;

#define T_DIM 65536
#define J_DIM 512
#define D_DIM 128
#define UT_STRIDE 520   // padded LDS row stride (halves): 1040B -> +4 banks/row, conflict-free b128 loads
#define BT 128          // t-columns per workgroup (8 waves x 16)

// out[d, t] = ( sum_j exp(s[t,j]-m[t]) * u[j,d] ) / Z[t]
// WMMA mapping: A = uT tile [16d x 32j] (from LDS), B = exp(s) tile [32j x 16t], C/D = [16d x 16t] f32.
__global__ __launch_bounds__(256)
void c2q_wmma_kernel(const float* __restrict__ u,
                     const float* __restrict__ s,
                     float* __restrict__ out)
{
    extern __shared__ char smem[];
    _Float16* uT = reinterpret_cast<_Float16*>(smem);   // [D_DIM][UT_STRIDE] f16

    const int tid  = threadIdx.x;
    const int lane = tid & 31;
    const int wave = tid >> 5;
    const int l15  = lane & 15;
    const int lh   = lane >> 4;          // half-wave select (K-halves of WMMA layouts)

    // ---- stage u^T into LDS as f16 (coalesced f32x4 reads, padded-row b16 writes) ----
    for (int idx = tid * 4; idx < J_DIM * D_DIM; idx += 256 * 4) {
        const float4 v = *reinterpret_cast<const float4*>(u + idx);
        const int j = idx >> 7;          // u row  (j)
        const int d = idx & 127;         // u col  (d) -> uT row
        uT[(d + 0) * UT_STRIDE + j] = (_Float16)v.x;
        uT[(d + 1) * UT_STRIDE + j] = (_Float16)v.y;
        uT[(d + 2) * UT_STRIDE + j] = (_Float16)v.z;
        uT[(d + 3) * UT_STRIDE + j] = (_Float16)v.w;
    }
    __syncthreads();

    const int    t_row = blockIdx.x * BT + wave * 16 + l15;   // this lane's t column
    const float* srow  = s + (size_t)t_row * J_DIM;

    // ---- pass 1: row max (each half-wave scans half the row, combine via xor-16) ----
    float m = -3.0e38f;
    {
        const float* p = srow + lh * (J_DIM / 2);
        #pragma unroll 8
        for (int i = 0; i < J_DIM / 2; i += 4) {
            const float4 v = *reinterpret_cast<const float4*>(p + i);
            m = fmaxf(m, fmaxf(fmaxf(v.x, v.y), fmaxf(v.z, v.w)));
        }
    }
    m = fmaxf(m, __shfl_xor(m, 16));

    // ---- main loop: build exp tile, accumulate Z, 8 WMMAs per 32-wide K step ----
    v8f acc[8];
    const v8f vzero = {};
    #pragma unroll
    for (int r = 0; r < 8; ++r) acc[r] = vzero;

    float zpart = 0.0f;

    for (int k0 = 0; k0 < J_DIM; k0 += 32) {
        // B fragment: lane L<16 -> K = k0..k0+15, lane L>=16 -> K = k0+16..k0+31 (contiguous order, matches A)
        const float* p  = srow + k0 + lh * 16;
        const float4 q0 = *reinterpret_cast<const float4*>(p + 0);
        const float4 q1 = *reinterpret_cast<const float4*>(p + 4);
        const float4 q2 = *reinterpret_cast<const float4*>(p + 8);
        const float4 q3 = *reinterpret_cast<const float4*>(p + 12);
        float se[16];
        se[0]=q0.x; se[1]=q0.y; se[2]=q0.z; se[3]=q0.w;
        se[4]=q1.x; se[5]=q1.y; se[6]=q1.z; se[7]=q1.w;
        se[8]=q2.x; se[9]=q2.y; se[10]=q2.z; se[11]=q2.w;
        se[12]=q3.x; se[13]=q3.y; se[14]=q3.z; se[15]=q3.w;

        v16h b;
        #pragma unroll
        for (int i = 0; i < 16; ++i) {
            const float ev = __expf(se[i] - m);   // unnormalized prob, <= 1.0, f16-safe
            zpart += ev;
            b[i] = (_Float16)ev;
        }

        // 8 d-tiles: A fragments from LDS.
        // Lane L<16: M=d row = 16r+L, K = k0..k0+15
        // Lane L>=16:            K = k0+16..k0+31   (contiguous, matches B)
        #pragma unroll
        for (int r = 0; r < 8; ++r) {
            const _Float16* arow = uT + (r * 16 + l15) * UT_STRIDE;
            const v8h a_lo = *reinterpret_cast<const v8h*>(arow + k0 +     lh * 16);
            const v8h a_hi = *reinterpret_cast<const v8h*>(arow + k0 + 8 + lh * 16);
            v16h a;
            #pragma unroll
            for (int i = 0; i < 8; ++i) { a[i] = a_lo[i]; a[i + 8] = a_hi[i]; }
            acc[r] = __builtin_amdgcn_wmma_f32_16x16x32_f16(
                         false, a, false, b, (short)0, acc[r], false, false);
        }
    }

    // ---- normalize by Z (per-t == per-lane-column of C/D tile) and store ----
    const float z   = zpart + __shfl_xor(zpart, 16);
    const float inv = 1.0f / z;

    // C/D layout: VGPR row v -> M = v (lanes 0-15) or v+8 (lanes 16-31); N = lane&15
    #pragma unroll
    for (int r = 0; r < 8; ++r) {
        #pragma unroll
        for (int row = 0; row < 8; ++row) {
            const int d = r * 16 + lh * 8 + row;
            out[(size_t)d * T_DIM + t_row] = acc[r][row] * inv;
        }
    }
}

extern "C" void kernel_launch(void* const* d_in, const int* in_sizes, int n_in,
                              void* d_out, int out_size, void* d_ws, size_t ws_size,
                              hipStream_t stream) {
    const float* u = (const float*)d_in[0];   // [1, 512, 128] f32
    const float* s = (const float*)d_in[1];   // [65536, 512]  f32
    float* out = (float*)d_out;               // [128, 65536]  f32

    const size_t lds_bytes = (size_t)D_DIM * UT_STRIDE * sizeof(_Float16); // 133,120 B (<320KB WGP LDS)
    hipFuncSetAttribute(reinterpret_cast<const void*>(c2q_wmma_kernel),
                        hipFuncAttributeMaxDynamicSharedMemorySize, (int)lds_bytes);

    dim3 grid(T_DIM / BT);   // 512 workgroups
    dim3 block(256);         // 8 waves (wave32)
    c2q_wmma_kernel<<<grid, block, lds_bytes, stream>>>(u, s, out);
}
